// NeuronLTX2Attention_65859028517586
// MI455X (gfx1250) — compile-verified, matrix operand order checked by reading, not run on hardware
//
#include <hip/hip_runtime.h>

// ---------------------------------------------------------------------------
// MI455X / gfx1250: wave32, WMMA bf16 16x16x32 f32-accumulate,
// async global->LDS staging (ASYNCcnt), flash attention.
// ---------------------------------------------------------------------------

typedef __attribute__((ext_vector_type(16))) __bf16 v16bf;
typedef __attribute__((ext_vector_type(8)))  __bf16 v8bf;
typedef __attribute__((ext_vector_type(8)))  float  v8f;

#define TOK_B 2
#define TOK_N 2048
#define DIM   4096
#define HEADS 32
#define HD    128
#define EPSF  1e-6f

// GEMM tiling
#define BM 128
#define BN 128
#define BK 32
#define LDK 40           // padded K stride (halves): 80 bytes, 16B aligned

// Attention tiling
#define LDH 136          // padded halves stride: 272 bytes, 16B aligned
#define LDS_S 132        // padded float stride for score buffer

static __device__ __forceinline__ v16bf combine16(v8bf lo, v8bf hi) {
    v16bf r;
#pragma unroll
    for (int i = 0; i < 8; ++i) { r[i] = lo[i]; r[8 + i] = hi[i]; }
    return r;
}

// generic pointer to __shared__ -> raw LDS byte offset (addr[31:0] per ISA 10.2)
static __device__ __forceinline__ unsigned lds_off_of(const void* p) {
    return (unsigned)(unsigned long long)p;
}

// 16-byte async copy global -> LDS, tracked by ASYNCcnt (ISA ch.8 §4)
static __device__ __forceinline__ void async_cp16(unsigned lds_byte_off, const void* gaddr) {
    asm volatile("global_load_async_to_lds_b128 %0, %1, off"
                 :: "v"(lds_byte_off), "v"(gaddr)
                 : "memory");
}

static __device__ __forceinline__ void async_wait0() {
    asm volatile("s_wait_asynccnt 0" ::: "memory");
}

// ---------------------------------------------------------------------------
// f32 -> bf16 conversion (vectorized)
// ---------------------------------------------------------------------------
__global__ __launch_bounds__(256) void cvt_f32_bf16(const float4* __restrict__ in,
                                                    __bf16* __restrict__ out, int n4) {
    int i = blockIdx.x * 256 + threadIdx.x;
    if (i < n4) {
        float4 v = in[i];
        __bf16* o = out + 4 * (size_t)i;
        o[0] = (__bf16)v.x; o[1] = (__bf16)v.y;
        o[2] = (__bf16)v.z; o[3] = (__bf16)v.w;
    }
}

// ---------------------------------------------------------------------------
// Tiled bf16 WMMA GEMM: C[M,Nout] = A[M,K] * B[K,Nout] + bias
// MODE 0: bias,             write bf16 to [B,H,N,HD] layout  (V projection)
// MODE 1: bias + RMSNorm*g, write bf16 to [B,H,N,HD] layout  (Q/K projection)
// MODE 2: bias,             write f32 row-major [M,Nout]     (O projection)
// Block = 256 threads = 8 waves (2x4). Wave tile = 64x32 = 4x2 WMMA tiles.
// ---------------------------------------------------------------------------
template <int MODE>
__global__ __launch_bounds__(256) void gemm_bf16(const __bf16* __restrict__ A,
                                                 const __bf16* __restrict__ Bmat,
                                                 const float* __restrict__ bias,
                                                 const float* __restrict__ g,
                                                 void* __restrict__ Out,
                                                 int M, int K, int Nout) {
    __shared__ __bf16 As[BM * LDK];
    __shared__ __bf16 Bs[BN * LDK];
    __shared__ float  rowss[BM];

    const int tid  = threadIdx.x;
    const int lane = tid & 31;
    const int wid  = tid >> 5;
    const int wm   = wid >> 2;   // 0..1
    const int wn   = wid & 3;    // 0..3
    const int row0 = blockIdx.y * BM;
    const int col0 = blockIdx.x * BN;

    // WMMA fragment lane mapping (ISA 7.12.2, 16-bit A 16x32)
    const int fm = lane & 15;
    const int kg = lane >> 4;
    const int o0 = kg * 8;          // halves 0..7  -> K o0..o0+7
    const int o1 = 16 + kg * 8;     // halves 8..15 -> K o1..o1+7

    v8f acc[4][2];
#pragma unroll
    for (int mt = 0; mt < 4; ++mt)
#pragma unroll
        for (int nt = 0; nt < 2; ++nt) acc[mt][nt] = (v8f)0.0f;

    // per-thread staging coordinates
    const int ar  = tid >> 1;            // A tile row
    const int acb = (tid & 1) * 16;      // A tile col base (halves)
    const unsigned a_lds = lds_off_of(As + ar * LDK + acb);

    for (int k0 = 0; k0 < K; k0 += BK) {
        // ---- stage A tile (128x32) via async global->LDS (ASYNCcnt path) ----
        {
            const __bf16* gp = A + (size_t)(row0 + ar) * K + k0 + acb;
            async_cp16(a_lds,      gp);
            async_cp16(a_lds + 16, gp + 8);
            if (k0 + BK < K)  // hint next tile toward L2 (global_prefetch_b8)
                __builtin_prefetch(gp + BK, 0, 1);
        }
        // ---- stage B tile (32x128) transposed -> Bs[n][k] (needs VGPR pass) ----
        {
            const int kk = tid >> 3;
            const int nb = (tid & 7) * 16;
            const v8bf* src = (const v8bf*)(Bmat + (size_t)(k0 + kk) * Nout + col0 + nb);
            v8bf x0 = src[0], x1 = src[1];
#pragma unroll
            for (int i = 0; i < 8; ++i) {
                Bs[(nb + i) * LDK + kk]     = x0[i];
                Bs[(nb + 8 + i) * LDK + kk] = x1[i];
            }
        }
        async_wait0();
        __syncthreads();

        v16bf bfrag[2];
#pragma unroll
        for (int nt = 0; nt < 2; ++nt) {
            const int n = wn * 32 + nt * 16 + fm;
            bfrag[nt] = combine16(*(const v8bf*)(Bs + n * LDK + o0),
                                  *(const v8bf*)(Bs + n * LDK + o1));
        }
#pragma unroll
        for (int mt = 0; mt < 4; ++mt) {
            const int m = wm * 64 + mt * 16 + fm;
            v16bf afrag = combine16(*(const v8bf*)(As + m * LDK + o0),
                                    *(const v8bf*)(As + m * LDK + o1));
#pragma unroll
            for (int nt = 0; nt < 2; ++nt)
                acc[mt][nt] = __builtin_amdgcn_wmma_f32_16x16x32_bf16(
                    false, afrag, false, bfrag[nt], (short)0, acc[mt][nt], false, false);
        }
        __syncthreads();
    }

    // ---- epilogue ----
    const int n_lane = lane & 15;
    const int mbase  = (lane >> 4) * 8;

#pragma unroll
    for (int nt = 0; nt < 2; ++nt) {
        const float bv = bias[col0 + wn * 32 + nt * 16 + n_lane];
#pragma unroll
        for (int mt = 0; mt < 4; ++mt)
#pragma unroll
            for (int r = 0; r < 8; ++r) acc[mt][nt][r] += bv;
    }

    if (MODE == 1) {
        if (tid < BM) rowss[tid] = 0.0f;
        __syncthreads();
#pragma unroll
        for (int mt = 0; mt < 4; ++mt)
#pragma unroll
            for (int nt = 0; nt < 2; ++nt)
#pragma unroll
                for (int r = 0; r < 8; ++r) {
                    const float y = acc[mt][nt][r];
                    atomicAdd(&rowss[wm * 64 + mt * 16 + mbase + r], y * y);
                }
        __syncthreads();
    }

#pragma unroll
    for (int mt = 0; mt < 4; ++mt)
#pragma unroll
        for (int nt = 0; nt < 2; ++nt) {
            const int col = col0 + wn * 32 + nt * 16 + n_lane;
#pragma unroll
            for (int r = 0; r < 8; ++r) {
                const int lr   = wm * 64 + mt * 16 + mbase + r;
                const int grow = row0 + lr;
                float y = acc[mt][nt][r];
                if (MODE == 1) {
                    const float ss = rowss[lr];
                    y = y * rsqrtf(ss * (1.0f / (float)HD) + EPSF) * g[col & (HD - 1)];
                }
                if (MODE <= 1) {
                    const int b    = grow >> 11;          // / TOK_N
                    const int nn   = grow & (TOK_N - 1);
                    const int head = col >> 7;            // / HD
                    const int hd   = col & (HD - 1);
                    ((__bf16*)Out)[(((size_t)b * HEADS + head) * TOK_N + nn) * HD + hd] =
                        (__bf16)y;
                } else {
                    ((float*)Out)[(size_t)grow * Nout + col] = y;
                }
            }
        }
}

// ---------------------------------------------------------------------------
// Flash attention per (qblock, head, batch). Q/K/V bf16 in [B,H,N,HD].
// Output bf16 in [B,N,H*HD] (row-major for the final GEMM).
// ---------------------------------------------------------------------------
__global__ __launch_bounds__(256) void attn_bf16(const __bf16* __restrict__ Q,
                                                 const __bf16* __restrict__ Kmat,
                                                 const __bf16* __restrict__ V,
                                                 __bf16* __restrict__ Out) {
    extern __shared__ char smem[];
    __bf16* Qs = (__bf16*)smem;                  // 128 x LDH
    __bf16* Ks = Qs + 128 * LDH;                 // 128 x LDH   (key-major: Ks[key][hd])
    __bf16* Vt = Ks + 128 * LDH;                 // 128 x LDH   (Vt[hd][key])
    float*  Sb = (float*)(Vt + 128 * LDH);       // 128 x LDS_S
    __bf16* Pb = (__bf16*)(Sb + 128 * LDS_S);    // 128 x LDH   (Pb[q][key])
    float*  mrow = (float*)(Pb + 128 * LDH);
    float*  lrow = mrow + 128;
    float*  crow = lrow + 128;

    const int tid  = threadIdx.x;
    const int lane = tid & 31;
    const int wid  = tid >> 5;
    const int wm   = wid >> 2;
    const int wn   = wid & 3;
    const int qb   = blockIdx.x;
    const int h    = blockIdx.y;
    const int b    = blockIdx.z;

    const int fm = lane & 15;
    const int kg = lane >> 4;
    const int o0 = kg * 8;
    const int o1 = 16 + kg * 8;
    const int n_lane = lane & 15;
    const int mbase  = (lane >> 4) * 8;

    const size_t headbase = ((size_t)b * HEADS + h) * TOK_N;
    const float scale = 0.08838834764831845f;     // 1/sqrt(128)

    // per-thread staging coordinates (contiguous 64-half runs)
    const int sr  = tid >> 1;
    const int scb = (tid & 1) * 64;
    const unsigned q_lds = lds_off_of(Qs + sr * LDH + scb);
    const unsigned k_lds = lds_off_of(Ks + sr * LDH + scb);

    // stage Q block via async global->LDS
    {
        const __bf16* gp = Q + (headbase + qb * 128 + sr) * HD + scb;
#pragma unroll
        for (int i = 0; i < 8; ++i)
            async_cp16(q_lds + i * 16, gp + i * 8);
    }
    if (tid < 128) { mrow[tid] = -1e30f; lrow[tid] = 0.0f; }

    v8f oacc[4][2];
#pragma unroll
    for (int mt = 0; mt < 4; ++mt)
#pragma unroll
        for (int nt = 0; nt < 2; ++nt) oacc[mt][nt] = (v8f)0.0f;

    for (int j = 0; j < TOK_N / 128; ++j) {
        // stage K block via async global->LDS: Ks[key][hd]
        {
            const __bf16* gp = Kmat + (headbase + j * 128 + sr) * HD + scb;
#pragma unroll
            for (int i = 0; i < 8; ++i)
                async_cp16(k_lds + i * 16, gp + i * 8);
        }
        // stage V block transposed: Vt[hd][key] (needs VGPR pass)
        {
            const int key = tid >> 1;
            const int hb  = (tid & 1) * 64;
            const v8bf* src = (const v8bf*)(V + (headbase + j * 128 + key) * HD + hb);
#pragma unroll
            for (int i = 0; i < 8; ++i) {
                v8bf x = src[i];
#pragma unroll
                for (int e = 0; e < 8; ++e)
                    Vt[(hb + i * 8 + e) * LDH + key] = x[e];
            }
        }
        async_wait0();                 // covers this K block (and Q on j==0)
        __syncthreads();

        // S = Q * K^T
        v8f sacc[4][2];
#pragma unroll
        for (int mt = 0; mt < 4; ++mt)
#pragma unroll
            for (int nt = 0; nt < 2; ++nt) sacc[mt][nt] = (v8f)0.0f;

#pragma unroll
        for (int kk = 0; kk < 4; ++kk) {
            const int ko = kk * 32;
            v16bf bfrag[2];
#pragma unroll
            for (int nt = 0; nt < 2; ++nt) {
                const int n = wn * 32 + nt * 16 + fm;
                bfrag[nt] = combine16(*(const v8bf*)(Ks + n * LDH + ko + o0),
                                      *(const v8bf*)(Ks + n * LDH + ko + o1));
            }
#pragma unroll
            for (int mt = 0; mt < 4; ++mt) {
                const int m = wm * 64 + mt * 16 + fm;
                v16bf afrag = combine16(*(const v8bf*)(Qs + m * LDH + ko + o0),
                                        *(const v8bf*)(Qs + m * LDH + ko + o1));
#pragma unroll
                for (int nt = 0; nt < 2; ++nt)
                    sacc[mt][nt] = __builtin_amdgcn_wmma_f32_16x16x32_bf16(
                        false, afrag, false, bfrag[nt], (short)0, sacc[mt][nt], false, false);
            }
        }
        // spill scaled scores to LDS
#pragma unroll
        for (int mt = 0; mt < 4; ++mt)
#pragma unroll
            for (int nt = 0; nt < 2; ++nt) {
                const int c = wn * 32 + nt * 16 + n_lane;
#pragma unroll
                for (int r = 0; r < 8; ++r) {
                    const int lr = wm * 64 + mt * 16 + mbase + r;
                    Sb[lr * LDS_S + c] = sacc[mt][nt][r] * scale;
                }
            }
        __syncthreads();

        // online softmax per query row
        if (tid < 128) {
            const float mo = mrow[tid];
            float bm = -1e30f;
            for (int c = 0; c < 128; ++c) bm = fmaxf(bm, Sb[tid * LDS_S + c]);
            const float mn = fmaxf(mo, bm);
            const float co = __expf(mo - mn);
            float s = 0.0f;
            for (int c = 0; c < 128; ++c) {
                const float e = __expf(Sb[tid * LDS_S + c] - mn);
                s += e;
                Pb[tid * LDH + c] = (__bf16)e;
            }
            mrow[tid] = mn;
            lrow[tid] = lrow[tid] * co + s;
            crow[tid] = co;
        }
        __syncthreads();

        // rescale running output, then O += P * V
#pragma unroll
        for (int mt = 0; mt < 4; ++mt)
#pragma unroll
            for (int r = 0; r < 8; ++r) {
                const float co = crow[wm * 64 + mt * 16 + mbase + r];
#pragma unroll
                for (int nt = 0; nt < 2; ++nt) oacc[mt][nt][r] *= co;
            }

#pragma unroll
        for (int kk = 0; kk < 4; ++kk) {
            const int ko = kk * 32;
            v16bf bfrag[2];
#pragma unroll
            for (int nt = 0; nt < 2; ++nt) {
                const int n = wn * 32 + nt * 16 + fm;   // n = hd column
                bfrag[nt] = combine16(*(const v8bf*)(Vt + n * LDH + ko + o0),
                                      *(const v8bf*)(Vt + n * LDH + ko + o1));
            }
#pragma unroll
            for (int mt = 0; mt < 4; ++mt) {
                const int m = wm * 64 + mt * 16 + fm;
                v16bf afrag = combine16(*(const v8bf*)(Pb + m * LDH + ko + o0),
                                        *(const v8bf*)(Pb + m * LDH + ko + o1));
#pragma unroll
                for (int nt = 0; nt < 2; ++nt)
                    oacc[mt][nt] = __builtin_amdgcn_wmma_f32_16x16x32_bf16(
                        false, afrag, false, bfrag[nt], (short)0, oacc[mt][nt], false, false);
            }
        }
        __syncthreads();
    }

    // normalize and write [B, N, H*HD] bf16
#pragma unroll
    for (int mt = 0; mt < 4; ++mt)
#pragma unroll
        for (int r = 0; r < 8; ++r) {
            const int lr = wm * 64 + mt * 16 + mbase + r;
            const float inv = 1.0f / lrow[lr];
            const size_t rowoff = ((size_t)b * TOK_N + qb * 128 + lr) * (HEADS * HD) + h * HD;
#pragma unroll
            for (int nt = 0; nt < 2; ++nt) {
                const int hd = wn * 32 + nt * 16 + n_lane;
                Out[rowoff + hd] = (__bf16)(oacc[mt][nt][r] * inv);
            }
        }
}

// ---------------------------------------------------------------------------
// Host launcher
// ---------------------------------------------------------------------------
extern "C" void kernel_launch(void* const* d_in, const int* in_sizes, int n_in,
                              void* d_out, int out_size, void* d_ws, size_t ws_size,
                              hipStream_t stream) {
    const float* X  = (const float*)d_in[0];
    const float* Wq = (const float*)d_in[1];
    const float* bq = (const float*)d_in[2];
    const float* Wk = (const float*)d_in[3];
    const float* bk = (const float*)d_in[4];
    const float* Wv = (const float*)d_in[5];
    const float* bv = (const float*)d_in[6];
    const float* Wo = (const float*)d_in[7];
    const float* bo = (const float*)d_in[8];
    const float* gq = (const float*)d_in[9];
    const float* gk = (const float*)d_in[10];

    const size_t SZ = (size_t)DIM * DIM;   // 16,777,216 elements (== B*N*D == B*H*N*HD)
    char* ws = (char*)d_ws;
    __bf16* Xb  = (__bf16*)ws; ws += SZ * 2;
    __bf16* Wqb = (__bf16*)ws; ws += SZ * 2;
    __bf16* Wkb = (__bf16*)ws; ws += SZ * 2;
    __bf16* Wvb = (__bf16*)ws; ws += SZ * 2;
    __bf16* Wob = (__bf16*)ws; ws += SZ * 2;
    __bf16* Qn  = (__bf16*)ws; ws += SZ * 2;
    __bf16* Kn  = (__bf16*)ws; ws += SZ * 2;
    __bf16* Vn  = (__bf16*)ws; ws += SZ * 2;
    __bf16* Ao  = (__bf16*)ws; ws += SZ * 2;

    const int n4 = (int)(SZ / 4);
    const int cblk = (n4 + 255) / 256;
    cvt_f32_bf16<<<cblk, 256, 0, stream>>>((const float4*)X,  Xb,  n4);
    cvt_f32_bf16<<<cblk, 256, 0, stream>>>((const float4*)Wq, Wqb, n4);
    cvt_f32_bf16<<<cblk, 256, 0, stream>>>((const float4*)Wk, Wkb, n4);
    cvt_f32_bf16<<<cblk, 256, 0, stream>>>((const float4*)Wv, Wvb, n4);
    cvt_f32_bf16<<<cblk, 256, 0, stream>>>((const float4*)Wo, Wob, n4);

    dim3 gg(DIM / BN, (TOK_B * TOK_N) / BM);   // 32 x 32
    dim3 blk(256);
    gemm_bf16<1><<<gg, blk, 0, stream>>>(Xb, Wqb, bq, gq, (void*)Qn, TOK_B * TOK_N, DIM, DIM);
    gemm_bf16<1><<<gg, blk, 0, stream>>>(Xb, Wkb, bk, gk, (void*)Kn, TOK_B * TOK_N, DIM, DIM);
    gemm_bf16<0><<<gg, blk, 0, stream>>>(Xb, Wvb, bv, nullptr, (void*)Vn, TOK_B * TOK_N, DIM, DIM);

    const size_t attn_lds = (size_t)(128 * LDH) * 2 * 4      // Qs, Ks, Vt, Pb
                          + (size_t)(128 * LDS_S) * 4        // Sb
                          + 3 * 128 * sizeof(float);         // mrow, lrow, crow
    attn_bf16<<<dim3(TOK_N / 128, HEADS, TOK_B), blk, attn_lds, stream>>>(Qn, Kn, Vn, Ao);

    gemm_bf16<2><<<gg, blk, 0, stream>>>(Ao, Wob, bo, nullptr, d_out, TOK_B * TOK_N, DIM, DIM);
}
